// SPTransformerEncoderLayer_80668075753648
// MI455X (gfx1250) — matrix operand order repaired and checked by reading, not compiled, not run on hardware
//
#include <hip/hip_runtime.h>
#include <hip/hip_bf16.h>

#define N_NODES 16384
#define E_EDGES 524288
#define DIM     128
#define HEADS   8
#define DHEAD   16
#define HIDDEN  512
#define BN_EPS  1e-5f

typedef __attribute__((ext_vector_type(16))) _Float16 v16h;
typedef __attribute__((ext_vector_type(8)))  float    v8f;

union V16H {
    uint4     u[2];
    v16h      h;
    _Float16  e[16];
};

// ---------------------------------------------------------------------------
// gfx1250 async global->LDS staging (ASYNCcnt path), with portable fallback
// ---------------------------------------------------------------------------
#if defined(__AMDGCN__) && __has_builtin(__builtin_amdgcn_global_load_async_to_lds_b128) && __has_builtin(__builtin_amdgcn_s_wait_asynccnt)
#define USE_ASYNC_LDS 1
#else
#define USE_ASYNC_LDS 0
#endif

#if USE_ASYNC_LDS
typedef __attribute__((__vector_size__(4 * sizeof(int)))) int vsi4;
typedef __attribute__((address_space(1))) vsi4* as1_i4p;   // global int4*
typedef __attribute__((address_space(3))) vsi4* as3_i4p;   // LDS int4*
#endif

__device__ __forceinline__ void lds_copy16(_Float16* dst, const _Float16* src) {
#if USE_ASYNC_LDS
    __builtin_amdgcn_global_load_async_to_lds_b128(
        (as1_i4p)(void*)src, (as3_i4p)(void*)dst, 0, 0);
#else
    *(uint4*)dst = *(const uint4*)src;
#endif
}
__device__ __forceinline__ void lds_copy_fence() {
#if USE_ASYNC_LDS
    __builtin_amdgcn_s_wait_asynccnt(0);
#endif
}

// ---------------------------------------------------------------------------
// elementwise conversions
// ---------------------------------------------------------------------------
__global__ void k_f32_to_f16(const float* __restrict__ in, _Float16* __restrict__ out, int n) {
    int i = blockIdx.x * 256 + threadIdx.x;
    if (i < n) out[i] = (_Float16)in[i];
}

// W [R,C] f32 (row-major)  ->  Wt [C,R] f16 (row-major)  (i.e. transposed)
__global__ void k_convT_f16(const float* __restrict__ W, _Float16* __restrict__ Wt, int R, int C) {
    int i = blockIdx.x * 256 + threadIdx.x;
    if (i < R * C) {
        int r = i / C, c = i % C;
        Wt[(size_t)c * R + r] = (_Float16)W[i];
    }
}

// ---------------------------------------------------------------------------
// WMMA GEMM:  C[M,Nout] = A[M,K] * B[K,Nout]   with Bt = B^T given as [Nout,K]
// block = 256 threads (8 waves), block tile = 128 x 64, wave tile = 32 x 32.
// K consumed in chunks of 128 staged via async global->LDS copies into a
// double buffer: chunk kc+128 streams in while WMMAs consume chunk kc.
// Optional fused residual / relu, f32 and/or f16 output.
// ---------------------------------------------------------------------------
#define LDA 136   // halves per LDS row (128 + 8 pad)
#define LDB 136

__global__ __launch_bounds__(256)
void k_gemm_wmma(const _Float16* __restrict__ A, const _Float16* __restrict__ Bt,
                 const float* __restrict__ resid, float* __restrict__ Cf,
                 _Float16* __restrict__ Ch, int M, int K, int Nout, int relu)
{
    __shared__ _Float16 sA[2][128 * LDA];
    __shared__ _Float16 sB[2][64 * LDB];

    const int tid  = threadIdx.x;
    const int lane = tid & 31;
    const int wid  = tid >> 5;
    const int wm   = wid & 3;        // wave row tile (0..3) -> rows 32*wm
    const int wn   = wid >> 2;       // wave col tile (0..1) -> cols 32*wn
    const int bm   = blockIdx.x;
    const int bn   = blockIdx.y;
    const int hi   = lane >> 4;      // 0 for lanes 0-15, 1 for lanes 16-31
    const int lml  = lane & 15;

    v8f zero = {0.f, 0.f, 0.f, 0.f, 0.f, 0.f, 0.f, 0.f};
    v8f acc[2][2];
    acc[0][0] = zero; acc[0][1] = zero; acc[1][0] = zero; acc[1][1] = zero;

    const int nch = K >> 7;          // number of 128-wide K chunks

    // ---- stage chunk 0 into buffer 0
    #pragma unroll
    for (int t = 0; t < 8; ++t) {
        int i = tid + t * 256, row = i >> 4, c = i & 15;
        lds_copy16(&sA[0][row * LDA + c * 8], A + (size_t)(bm * 128 + row) * K + c * 8);
    }
    #pragma unroll
    for (int t = 0; t < 4; ++t) {
        int i = tid + t * 256, row = i >> 4, c = i & 15;
        lds_copy16(&sB[0][row * LDB + c * 8], Bt + (size_t)(bn * 64 + row) * K + c * 8);
    }
    lds_copy_fence();
    __syncthreads();

    int buf = 0;
    for (int ch = 0; ch < nch; ++ch) {
        // ---- prefetch next chunk into the other buffer (async, no wait yet)
        if (ch + 1 < nch) {
            int kn = (ch + 1) << 7;
            #pragma unroll
            for (int t = 0; t < 8; ++t) {
                int i = tid + t * 256, row = i >> 4, c = i & 15;
                lds_copy16(&sA[buf ^ 1][row * LDA + c * 8],
                           A + (size_t)(bm * 128 + row) * K + kn + c * 8);
            }
            #pragma unroll
            for (int t = 0; t < 4; ++t) {
                int i = tid + t * 256, row = i >> 4, c = i & 15;
                lds_copy16(&sB[buf ^ 1][row * LDB + c * 8],
                           Bt + (size_t)(bn * 64 + row) * K + kn + c * 8);
            }
        }

        // ---- consume current buffer: 4 K-steps of 2x2 WMMA tiles
        const _Float16* pA = sA[buf];
        const _Float16* pB = sB[buf];
        #pragma unroll
        for (int k0 = 0; k0 < 128; k0 += 32) {
            V16H a[2], b[2];
            // A fragment 16x32 f16 (ISA 7.12.2): lanes<16 hold K = k0+{0..7,16..23},
            // lanes>=16 hold K = k0+{8..15,24..31}; two b128 LDS loads each.
            #pragma unroll
            for (int i = 0; i < 2; ++i) {
                int row = wm * 32 + i * 16 + lml;
                int kof = k0 + hi * 8;
                a[i].u[0] = *(const uint4*)(&pA[row * LDA + kof]);
                a[i].u[1] = *(const uint4*)(&pA[row * LDA + kof + 16]);
            }
            // B fragment 32x16: lane holds column, lanes<16 K=k0+{0..7,16..23},
            // lanes>=16 K=k0+{8..15,24..31}; same K order as A.
            #pragma unroll
            for (int j = 0; j < 2; ++j) {
                int col = wn * 32 + j * 16 + lml;
                int kof = k0 + hi * 8;
                b[j].u[0] = *(const uint4*)(&pB[col * LDB + kof]);
                b[j].u[1] = *(const uint4*)(&pB[col * LDB + kof + 16]);
            }
            #pragma unroll
            for (int i = 0; i < 2; ++i)
                #pragma unroll
                for (int j = 0; j < 2; ++j)
                    acc[i][j] = __builtin_amdgcn_wmma_f32_16x16x32_f16(
                        false, a[i].h, false, b[j].h, (short)0, acc[i][j], false, false);
        }

        if (ch + 1 < nch) {
            lds_copy_fence();   // own async copies landed
            __syncthreads();    // everyone's copies landed / reads done
            buf ^= 1;
        }
    }

    // ---- epilogue: C/D layout -> row = r + 8*hi, col = lane&15 within tile
    #pragma unroll
    for (int i = 0; i < 2; ++i)
        #pragma unroll
        for (int j = 0; j < 2; ++j)
            #pragma unroll
            for (int r = 0; r < 8; ++r) {
                int row = bm * 128 + wm * 32 + i * 16 + hi * 8 + r;
                int col = bn * 64  + wn * 32 + j * 16 + lml;
                float v = acc[i][j][r];
                if (resid) v += resid[(size_t)row * Nout + col];
                if (relu)  v = fmaxf(v, 0.0f);
                if (Cf) Cf[(size_t)row * Nout + col] = v;
                if (Ch) Ch[(size_t)row * Nout + col] = (_Float16)v;
            }
}

// ---------------------------------------------------------------------------
// edge-parallel attention pieces
// ---------------------------------------------------------------------------
__device__ __forceinline__ unsigned f2mono(float f) {
    unsigned u = __float_as_uint(f);
    return (u & 0x80000000u) ? ~u : (u | 0x80000000u);
}
__device__ __forceinline__ float mono2f(unsigned m) {
    return __uint_as_float((m & 0x80000000u) ? (m ^ 0x80000000u) : ~m);
}

__global__ void k_edge_scores(const _Float16* __restrict__ qh, const _Float16* __restrict__ kh,
                              const int* __restrict__ src, const int* __restrict__ dst,
                              float* __restrict__ scores, unsigned* __restrict__ mbits)
{
    int idx = blockIdx.x * 256 + threadIdx.x;
    if (idx >= E_EDGES * HEADS) return;
    int e = idx >> 3, h = idx & 7;
    int d = dst[e], s = src[e];
    V16H qv, kv;
    qv.u[0] = *(const uint4*)(qh + (size_t)d * DIM + h * DHEAD);
    qv.u[1] = *(const uint4*)(qh + (size_t)d * DIM + h * DHEAD + 8);
    kv.u[0] = *(const uint4*)(kh + (size_t)s * DIM + h * DHEAD);
    kv.u[1] = *(const uint4*)(kh + (size_t)s * DIM + h * DHEAD + 8);
    float sum = 0.f;
    #pragma unroll
    for (int i = 0; i < 16; ++i) sum += (float)qv.e[i] * (float)kv.e[i];
    sum *= 0.25f;                      // 1/sqrt(DH), DH=16
    scores[idx] = sum;
    atomicMax(&mbits[d * HEADS + h], f2mono(sum));
}

__global__ void k_alpha(const int* __restrict__ dst, float* __restrict__ scores,
                        const unsigned* __restrict__ mbits, float* __restrict__ denom)
{
    int idx = blockIdx.x * 256 + threadIdx.x;
    if (idx >= E_EDGES * HEADS) return;
    int e = idx >> 3, h = idx & 7;
    int d = dst[e];
    float m = mono2f(mbits[d * HEADS + h]);
    float a = __expf(scores[idx] - m);
    scores[idx] = a;                   // alpha overwrites score in place
    atomicAdd(&denom[d * HEADS + h], a);
}

__global__ void k_aggregate(const int* __restrict__ src, const int* __restrict__ dst,
                            const float* __restrict__ alpha, const float* __restrict__ denom,
                            const _Float16* __restrict__ vh, float* __restrict__ agg)
{
    int idx = blockIdx.x * 256 + threadIdx.x;
    if (idx >= E_EDGES * HEADS) return;
    int e = idx >> 3, h = idx & 7;
    int d = dst[e], s = src[e];
    float w = alpha[idx] / fmaxf(denom[d * HEADS + h], 1e-9f);
    V16H vv;
    vv.u[0] = *(const uint4*)(vh + (size_t)s * DIM + h * DHEAD);
    vv.u[1] = *(const uint4*)(vh + (size_t)s * DIM + h * DHEAD + 8);
    float* ap = agg + (size_t)d * DIM + h * DHEAD;
    #pragma unroll
    for (int i = 0; i < 16; ++i) atomicAdd(&ap[i], w * (float)vv.e[i]);
}

// ---------------------------------------------------------------------------
// batch-norm over the node axis (per column)
// ---------------------------------------------------------------------------
__global__ void k_colstats(const float* __restrict__ X, float* __restrict__ stats)
{
    __shared__ float ssum[256];
    __shared__ float ssq[256];
    int c = blockIdx.x, tid = threadIdx.x;
    float s = 0.f, q = 0.f;
    for (int r = tid; r < N_NODES; r += 256) {
        float v = X[(size_t)r * DIM + c];
        s += v; q += v * v;
    }
    ssum[tid] = s; ssq[tid] = q;
    __syncthreads();
    for (int st = 128; st > 0; st >>= 1) {
        if (tid < st) { ssum[tid] += ssum[tid + st]; ssq[tid] += ssq[tid + st]; }
        __syncthreads();
    }
    if (tid == 0) {
        float mu = ssum[0] / (float)N_NODES;
        stats[c]       = mu;
        stats[DIM + c] = ssq[0] / (float)N_NODES - mu * mu;
    }
}

__global__ void k_bn(const float* __restrict__ X, const float* __restrict__ stats,
                     const float* __restrict__ gamma, const float* __restrict__ beta,
                     float* __restrict__ outF, _Float16* __restrict__ outH)
{
    int i = blockIdx.x * 256 + threadIdx.x;
    if (i >= N_NODES * DIM) return;
    int c = i & (DIM - 1);
    float v = gamma[c] * (X[i] - stats[c]) * rsqrtf(stats[DIM + c] + BN_EPS) + beta[c];
    if (outF) outF[i] = v;
    if (outH) outH[i] = (_Float16)v;
}

// ---------------------------------------------------------------------------
// launcher
// ---------------------------------------------------------------------------
extern "C" void kernel_launch(void* const* d_in, const int* in_sizes, int n_in,
                              void* d_out, int out_size, void* d_ws, size_t ws_size,
                              hipStream_t stream)
{
    (void)in_sizes; (void)n_in; (void)out_size; (void)ws_size;

    const float* x    = (const float*)d_in[0];
    const int*   eidx = (const int*)d_in[1];
    const int*   src  = eidx;              // edge_index[0]
    const int*   dst  = eidx + E_EDGES;    // edge_index[1]
    const float* Wq = (const float*)d_in[2];
    const float* Wk = (const float*)d_in[3];
    const float* Wv = (const float*)d_in[4];
    const float* Wo = (const float*)d_in[5];
    const float* g1 = (const float*)d_in[6];
    const float* b1 = (const float*)d_in[7];
    const float* W1 = (const float*)d_in[8];
    const float* W2 = (const float*)d_in[9];
    const float* g2 = (const float*)d_in[10];
    const float* b2 = (const float*)d_in[11];

    char* base = (char*)d_ws;
    size_t off = 0;
    auto alloc = [&](size_t bytes) -> void* {
        void* p = base + off;
        off += (bytes + 255) & ~(size_t)255;
        return p;
    };

    _Float16* xh    = (_Float16*)alloc((size_t)N_NODES * DIM * 2);
    _Float16* Wqt   = (_Float16*)alloc((size_t)DIM * DIM * 2);
    _Float16* Wkt   = (_Float16*)alloc((size_t)DIM * DIM * 2);
    _Float16* Wvt   = (_Float16*)alloc((size_t)DIM * DIM * 2);
    _Float16* Wot   = (_Float16*)alloc((size_t)DIM * DIM * 2);
    _Float16* W1t   = (_Float16*)alloc((size_t)DIM * HIDDEN * 2);
    _Float16* W2t   = (_Float16*)alloc((size_t)HIDDEN * DIM * 2);
    _Float16* qh    = (_Float16*)alloc((size_t)N_NODES * DIM * 2);
    _Float16* kh    = (_Float16*)alloc((size_t)N_NODES * DIM * 2);
    _Float16* vh    = (_Float16*)alloc((size_t)N_NODES * DIM * 2);
    float*    scores= (float*)alloc((size_t)E_EDGES * HEADS * 4);   // alpha in place
    unsigned* mbits = (unsigned*)alloc((size_t)N_NODES * HEADS * 4);
    float*    denom = (float*)alloc((size_t)N_NODES * HEADS * 4);
    float*    agg   = (float*)alloc((size_t)N_NODES * DIM * 4);
    float*    r1    = (float*)alloc((size_t)N_NODES * DIM * 4);
    float*    x1    = (float*)alloc((size_t)N_NODES * DIM * 4);
    _Float16* hh    = (_Float16*)alloc((size_t)N_NODES * HIDDEN * 2);
    float*    stats1= (float*)alloc((size_t)2 * DIM * 4);
    float*    stats2= (float*)alloc((size_t)2 * DIM * 4);
    // reuse dead buffers
    _Float16* aggh  = qh;   // qh dead after alpha pass
    _Float16* x1h   = kh;   // kh dead after alpha pass
    float*    r2    = agg;  // agg dead after aggh conversion

    const int ND  = N_NODES * DIM;
    const int EH  = E_EDGES * HEADS;
    dim3 blk(256);

    // zero the atomic targets every call (ws is poisoned / reused)
    (void)hipMemsetAsync(mbits, 0, (size_t)N_NODES * HEADS * 4, stream);
    (void)hipMemsetAsync(denom, 0, (size_t)N_NODES * HEADS * 4, stream);
    (void)hipMemsetAsync(agg,   0, (size_t)N_NODES * DIM * 4, stream);

    // f16 conversions (+ weight transposes)
    k_f32_to_f16<<<(ND + 255) / 256, blk, 0, stream>>>(x, xh, ND);
    k_convT_f16<<<(DIM * DIM + 255) / 256, blk, 0, stream>>>(Wq, Wqt, DIM, DIM);
    k_convT_f16<<<(DIM * DIM + 255) / 256, blk, 0, stream>>>(Wk, Wkt, DIM, DIM);
    k_convT_f16<<<(DIM * DIM + 255) / 256, blk, 0, stream>>>(Wv, Wvt, DIM, DIM);
    k_convT_f16<<<(DIM * DIM + 255) / 256, blk, 0, stream>>>(Wo, Wot, DIM, DIM);
    k_convT_f16<<<(DIM * HIDDEN + 255) / 256, blk, 0, stream>>>(W1, W1t, DIM, HIDDEN);
    k_convT_f16<<<(HIDDEN * DIM + 255) / 256, blk, 0, stream>>>(W2, W2t, HIDDEN, DIM);

    // q, k, v projections (WMMA)
    dim3 gproj(N_NODES / 128, DIM / 64);
    k_gemm_wmma<<<gproj, blk, 0, stream>>>(xh, Wqt, nullptr, nullptr, qh, N_NODES, DIM, DIM, 0);
    k_gemm_wmma<<<gproj, blk, 0, stream>>>(xh, Wkt, nullptr, nullptr, kh, N_NODES, DIM, DIM, 0);
    k_gemm_wmma<<<gproj, blk, 0, stream>>>(xh, Wvt, nullptr, nullptr, vh, N_NODES, DIM, DIM, 0);

    // segment softmax attention over edges
    int egrid = (EH + 255) / 256;
    k_edge_scores<<<egrid, blk, 0, stream>>>(qh, kh, src, dst, scores, mbits);
    k_alpha<<<egrid, blk, 0, stream>>>(dst, scores, mbits, denom);
    k_aggregate<<<egrid, blk, 0, stream>>>(src, dst, scores, denom, vh, agg);

    // attn = agg @ Wo + x  (residual fused)
    k_f32_to_f16<<<(ND + 255) / 256, blk, 0, stream>>>(agg, aggh, ND);
    k_gemm_wmma<<<gproj, blk, 0, stream>>>(aggh, Wot, x, r1, nullptr, N_NODES, DIM, DIM, 0);

    // norm1
    k_colstats<<<DIM, blk, 0, stream>>>(r1, stats1);
    k_bn<<<(ND + 255) / 256, blk, 0, stream>>>(r1, stats1, g1, b1, x1, x1h);

    // FFN: relu(x1@W1) @ W2 + x1 (relu + residual fused into GEMM epilogues)
    dim3 gffn1(N_NODES / 128, HIDDEN / 64);
    k_gemm_wmma<<<gffn1, blk, 0, stream>>>(x1h, W1t, nullptr, nullptr, hh, N_NODES, DIM, HIDDEN, 1);
    k_gemm_wmma<<<gproj, blk, 0, stream>>>(hh, W2t, x1, r2, nullptr, N_NODES, HIDDEN, DIM, 0);

    // norm2 -> final output (f32)
    k_colstats<<<DIM, blk, 0, stream>>>(r2, stats2);
    k_bn<<<(ND + 255) / 256, blk, 0, stream>>>(r2, stats2, g2, b2, (float*)d_out, nullptr);
}
